// QKVAttention_16088947491212
// MI455X (gfx1250) — compile-verified, matrix operand order checked by reading, not run on hardware
//
#include <hip/hip_runtime.h>

// Reference: full cross-head attention, B=2, H=12, S=512, D=64 (fp32).
//   score[b,i,j,s,t] = (Q[b,i,s,:] . K[b,j,t,:]) / 8
//   W = softmax_t(score);  ctx[b,i,s,:] = sum_j W[b,i,j,s,:] @ V[b,j]
// Outputs concatenated in d_out: context [B,S,H*D] f32, then W [B,H,H,S,S] f32.
// valid_lens (d_in[3]) is accepted but unused by the reference math.
//
// MI455X roofline: weights tensor = 302 MB of f32 writes -> HBM floor ~13us at
// 23.3 TB/s; FLOPs ~19.3G are cheap on the f16 WMMA path. So: f16 inputs with
// f32 accumulate (v_wmma_f32_16x16x32_f16), f32 softmax, b128 LDS/global
// traffic everywhere, and a fully parallel vectorized softmax so the (low
// occupancy, LDS-bound) blocks don't serialize on it.

typedef __attribute__((ext_vector_type(16))) _Float16 v16h;
typedef __attribute__((ext_vector_type(8)))  _Float16 v8h;
typedef __attribute__((ext_vector_type(8)))  float    v8f;

#define B_      2
#define H_      12
#define S_      512
#define D_      64
#define VSTRIDE 520   // padded t-stride (halves) for transposed V in LDS

#define WMMA_F16(A, Bm, C) \
  __builtin_amdgcn_wmma_f32_16x16x32_f16(false, (A), false, (Bm), (short)0, (C), false, false)

// LDS layout (bytes):
//   sS   : 4*16*512*4 = 131072   f32 scores -> exp values
//   sP   : 4*16*512*2 =  65536   f16 normalized probs (WMMA A-fragments)
//   sKV  : 64*520*2   =  66560   f16 K (row-major [t][d]) then V (transposed [d][t])
//   sInv : 64*4       =    256   per-row 1/sum
#define SS_OFF   0
#define SP_OFF   (4 * 16 * S_ * 4)
#define SKV_OFF  (SP_OFF + 4 * 16 * S_ * 2)
#define SINV_OFF (SKV_OFF + 64 * VSTRIDE * 2)
#define LDS_BYTES (SINV_OFF + 64 * 4)

__global__ __launch_bounds__(128) void qkv_crosshead_attn_kernel(
    const float* __restrict__ Q, const float* __restrict__ K,
    const float* __restrict__ V, float* __restrict__ out_ctx,
    float* __restrict__ out_w)
{
  extern __shared__ char smem[];
  float*    sS   = (float*)(smem + SS_OFF);
  _Float16* sP   = (_Float16*)(smem + SP_OFF);
  _Float16* sKV  = (_Float16*)(smem + SKV_OFF);
  float*    sInv = (float*)(smem + SINV_OFF);

  const int tid  = threadIdx.x;
  const int lane = tid & 31;
  const int wv   = tid >> 5;            // wave 0..3
  const int tile = blockIdx.x & 7;      // 8 tiles of 64 query rows
  const int i    = (blockIdx.x >> 3) % H_;
  const int b    = blockIdx.x / (8 * H_);
  const int m0   = tile * 64 + wv * 16; // first query row owned by this wave
  const int lg   = lane >> 4;           // lane group (0/1) per ISA A/B layout
  const int ln   = lane & 15;

  // ---- A fragments for Q (f32 -> f16), built once, reused for every kv head ----
  // 16-bit A 16x32 layout: lane m=ln; K = {lg*8..+7, lg*8+16..+23} per 32-chunk.
  const float* qrow = Q + ((size_t)((b * H_ + i) * S_) + m0 + ln) * D_;
  v16h a0, a1;
#pragma unroll
  for (int h = 0; h < 8; ++h) {
    a0[h]     = (_Float16)qrow[lg * 8 + h];
    a0[h + 8] = (_Float16)qrow[lg * 8 + 16 + h];
    a1[h]     = (_Float16)qrow[32 + lg * 8 + h];
    a1[h + 8] = (_Float16)qrow[32 + lg * 8 + 16 + h];
  }

  const v8f vzero = {};
  v8f acc[4];                                 // 16x64 f32 context tile per wave
#pragma unroll
  for (int dt = 0; dt < 4; ++dt) acc[dt] = vzero;

  float*    srow = sS + (wv * 16) * S_;       // this wave's 16x512 score block
  _Float16* sPw  = sP + (wv * 16) * S_;       // this wave's 16x512 f16 probs

  for (int j = 0; j < H_; ++j) {
    __syncthreads();
    // ---- stage K_j into LDS as f16, row-major [t][d] ----
    {
      const float4* ksrc = (const float4*)(K + (size_t)((b * H_ + j) * S_) * D_);
      for (int k = 0; k < 64; ++k) {
        int e4 = k * 128 + tid;
        float4 v = ksrc[e4];
        int e = e4 * 4;
        sKV[e]     = (_Float16)v.x; sKV[e + 1] = (_Float16)v.y;
        sKV[e + 2] = (_Float16)v.z; sKV[e + 3] = (_Float16)v.w;
      }
    }
    __syncthreads();

    // ---- scores: 32 tiles of 16x16, each = 2x v_wmma_f32_16x16x32_f16 over d ----
    for (int tt = 0; tt < 32; ++tt) {
      int t = tt * 16 + ln;                   // B column n == key position t
      const v8h* kb = (const v8h*)(sKV + t * D_ + lg * 8); // 16B-aligned
      v8h k00 = kb[0], k01 = kb[2], k10 = kb[4], k11 = kb[6]; // d: +0,+16,+32,+48
      v16h b0 = __builtin_shufflevector(k00, k01, 0,1,2,3,4,5,6,7,8,9,10,11,12,13,14,15);
      v16h b1 = __builtin_shufflevector(k10, k11, 0,1,2,3,4,5,6,7,8,9,10,11,12,13,14,15);
      v8f c = vzero;
      c = WMMA_F16(a0, b0, c);
      c = WMMA_F16(a1, b1, c);
#pragma unroll
      for (int r = 0; r < 8; ++r)             // C: VGPR r -> M = r + 8*lg, N = ln
        srow[(r + lg * 8) * S_ + t] = c[r] * 0.125f;   // /sqrt(64)
    }
    __syncthreads();

    // ---- row softmax over t: 2 lanes per row (halves), float4-vectorized ----
    {
      float* r = srow + ln * S_ + lg * 256;   // this lane's half-row
      const float4* r4 = (const float4*)r;
      float4* w4 = (float4*)r;
      float mx = -3.402823466e38f;
      for (int t = 0; t < 64; ++t) {
        float4 v = r4[t];
        mx = fmaxf(mx, fmaxf(fmaxf(v.x, v.y), fmaxf(v.z, v.w)));
      }
      mx = fmaxf(mx, __shfl_xor(mx, 16, 32)); // combine the two half-rows
      float sum = 0.0f;
      for (int t = 0; t < 64; ++t) {
        float4 v = r4[t];
        v.x = __expf(v.x - mx); v.y = __expf(v.y - mx);
        v.z = __expf(v.z - mx); v.w = __expf(v.w - mx);
        sum += (v.x + v.y) + (v.z + v.w);
        w4[t] = v;                            // srow now holds exp values
      }
      sum += __shfl_xor(sum, 16, 32);
      float inv = 1.0f / sum;
      if (lg == 0) sInv[wv * 16 + ln] = inv;
      // f16 normalized probs -> sP (A-fragments for P@V), packed b128 stores
      v8h* p8 = (v8h*)(sPw + ln * S_ + lg * 256);
      for (int t = 0; t < 32; ++t) {
        float4 u = r4[t * 2], v = r4[t * 2 + 1];
        v8h o;
        o[0] = (_Float16)(u.x * inv); o[1] = (_Float16)(u.y * inv);
        o[2] = (_Float16)(u.z * inv); o[3] = (_Float16)(u.w * inv);
        o[4] = (_Float16)(v.x * inv); o[5] = (_Float16)(v.y * inv);
        o[6] = (_Float16)(v.z * inv); o[7] = (_Float16)(v.w * inv);
        p8[t] = o;
      }
    }
    __syncthreads();

    // ---- coalesced float4 write of attention_weights[b][i][j][m0..m0+15][:] ----
    {
      float4* wdst = (float4*)(out_w + (((size_t)(b * H_ + i) * H_ + j) * S_ + m0) * S_);
      const float4* s4 = (const float4*)srow;
      for (int k = 0; k < 64; ++k) {
        int idx4 = k * 32 + lane;             // float4 index, linear over 16x512
        float inv = sInv[wv * 16 + (idx4 >> 7)];
        float4 v = s4[idx4];
        v.x *= inv; v.y *= inv; v.z *= inv; v.w *= inv;
        wdst[idx4] = v;
      }
    }
    __syncthreads();

    // ---- stage V_j transposed into LDS: [d][t], stride 520 halves (bank-safe) ----
    {
      const float4* vsrc = (const float4*)(V + (size_t)((b * H_ + j) * S_) * D_);
      for (int k = 0; k < 64; ++k) {
        int e4 = k * 128 + tid;
        float4 v = vsrc[e4];
        int e = e4 * 4;
        int t = e >> 6, d = e & 63;
        sKV[(d + 0) * VSTRIDE + t] = (_Float16)v.x;
        sKV[(d + 1) * VSTRIDE + t] = (_Float16)v.y;
        sKV[(d + 2) * VSTRIDE + t] = (_Float16)v.z;
        sKV[(d + 3) * VSTRIDE + t] = (_Float16)v.w;
      }
    }
    __syncthreads();

    // ---- ctx += P @ V_j : K-dim is t (16 chunks of 32), N-dim is d (4 tiles) ----
    for (int c = 0; c < 16; ++c) {
      const v8h* pb = (const v8h*)(sPw + ln * S_ + c * 32 + lg * 8); // 16B-aligned
      v8h p0 = pb[0], p1 = pb[2];             // t: +0, +16
      v16h ap = __builtin_shufflevector(p0, p1, 0,1,2,3,4,5,6,7,8,9,10,11,12,13,14,15);
#pragma unroll
      for (int dt = 0; dt < 4; ++dt) {
        int dcol = dt * 16 + ln;              // B column n == d
        const v8h* vb = (const v8h*)(sKV + dcol * VSTRIDE + c * 32 + lg * 8);
        v8h v0 = vb[0], v1 = vb[2];           // t: +0, +16
        v16h bv = __builtin_shufflevector(v0, v1, 0,1,2,3,4,5,6,7,8,9,10,11,12,13,14,15);
        acc[dt] = WMMA_F16(ap, bv, acc[dt]);
      }
    }
  }

  // ---- write context[b, s, i*64 + d] ----
  float* obase = out_ctx + ((size_t)b * S_ + m0) * (H_ * D_) + i * D_;
#pragma unroll
  for (int dt = 0; dt < 4; ++dt)
#pragma unroll
    for (int r = 0; r < 8; ++r) {
      int m = r + lg * 8;
      obase[(size_t)m * (H_ * D_) + dt * 16 + ln] = acc[dt][r];
    }
}

extern "C" void kernel_launch(void* const* d_in, const int* in_sizes, int n_in,
                              void* d_out, int out_size, void* d_ws, size_t ws_size,
                              hipStream_t stream) {
  const float* Q = (const float*)d_in[0];
  const float* K = (const float*)d_in[1];
  const float* V = (const float*)d_in[2];
  // d_in[3] = valid_lens: unused by the reference computation.
  float* ctx = (float*)d_out;                               // [B,S,H*D]
  float* wts = ctx + (size_t)B_ * S_ * H_ * D_;             // [B,H,H,S,S]

  dim3 grid(B_ * H_ * (S_ / 64));
  dim3 block(128);
  qkv_crosshead_attn_kernel<<<grid, block, (size_t)LDS_BYTES, stream>>>(Q, K, V, ctx, wts);
}
